// TransformerDo_57715770524181
// MI455X (gfx1250) — matrix operand order repaired and checked by reading, not compiled, not run on hardware
//
#include <hip/hip_runtime.h>

// ---------------------------------------------------------------------------
// Types for CDNA5 WMMA (wave32): v_wmma_f32_16x16x32_bf16
// ---------------------------------------------------------------------------
typedef __attribute__((ext_vector_type(16))) __bf16 v16bf;
typedef __attribute__((ext_vector_type(8)))  __bf16 bf16x8;
typedef __attribute__((ext_vector_type(8)))  float  v8f;
typedef __attribute__((__vector_size__(16))) int          vint4;
typedef __attribute__((__vector_size__(16))) unsigned int vuint4;
typedef __attribute__((__vector_size__(32))) int          vint8;

#define WMMA_BF16(a, b, c) \
    __builtin_amdgcn_wmma_f32_16x16x32_bf16(false, (a), false, (b), (short)0, (c), false, false)

// gfx1250 async global->LDS path (ASYNCcnt), guarded so absence still compiles.
#if defined(__HIP_DEVICE_COMPILE__) && \
    __has_builtin(__builtin_amdgcn_global_load_async_to_lds_b128) && \
    __has_builtin(__builtin_amdgcn_s_wait_asynccnt)
#define USE_ASYNC_LDS 1
#else
#define USE_ASYNC_LDS 0
#endif

// gfx1250 Tensor Data Mover (TENSORcnt); this toolchain uses the 6-arg form:
// (uint32x4 g0, int32x8 g1, int32x4 g2, int32x4 g3, int32x8, i32 cpol)
#if defined(__HIP_DEVICE_COMPILE__) && \
    __has_builtin(__builtin_amdgcn_tensor_load_to_lds) && \
    __has_builtin(__builtin_amdgcn_s_wait_tensorcnt)
#define USE_TDM 1
#else
#define USE_TDM 0
#endif

#define GLOBAL_AS __attribute__((address_space(1)))
#define LDS_AS    __attribute__((address_space(3)))

// Model dims (fixed by the reference)
#define BB 2
#define LL 1024
#define DD 1024
#define HH 16
#define NL 8
#define VV 32000
#define FF 4096
#define DH 64
#define MM (BB*LL)   // 2048 token rows

// ---------------------------------------------------------------------------
// Fragment loaders (ISA 7.12.2 layouts).
// ---------------------------------------------------------------------------
__device__ __forceinline__ v16bf load_fragA(const __bf16* rowp, int hi) {
    bf16x8 lo = *(const bf16x8*)(rowp + 8 * hi);
    bf16x8 up = *(const bf16x8*)(rowp + 16 + 8 * hi);
    v16bf f;
#pragma unroll
    for (int e = 0; e < 8; ++e) { f[e] = lo[e]; f[e + 8] = up[e]; }
    return f;
}

__device__ __forceinline__ v16bf load_fragB(const __bf16* colp, int hi) {
    bf16x8 lo = *(const bf16x8*)(colp + 8 * hi);
    bf16x8 up = *(const bf16x8*)(colp + 16 + 8 * hi);
    v16bf f;
#pragma unroll
    for (int e = 0; e < 8; ++e) { f[e] = lo[e]; f[e + 8] = up[e]; }
    return f;
}

// ---------------------------------------------------------------------------
// Weight pre-pass: W f32 [K][N] -> Wt bf16 [N][K]  (64x64 tiles via LDS)
// ---------------------------------------------------------------------------
__global__ __launch_bounds__(256) void transp_cvt_kernel(
    const float* __restrict__ W, __bf16* __restrict__ Wt, int Kd, int Nd)
{
    __shared__ float tile[64][65];
    const int nt = blockIdx.x * 64;
    const int kt = blockIdx.y * 64;
    const int tx = threadIdx.x & 63;
    const int ty = threadIdx.x >> 6;   // 0..3
#pragma unroll
    for (int j = 0; j < 16; ++j) {
        int k = ty * 16 + j;
        tile[k][tx] = W[(size_t)(kt + k) * Nd + nt + tx];
    }
    __syncthreads();
#pragma unroll
    for (int j = 0; j < 16; ++j) {
        int n = ty * 16 + j;
        Wt[(size_t)(nt + n) * Kd + kt + tx] = (__bf16)tile[tx][n];
    }
}

// ---------------------------------------------------------------------------
// GEMM:  C[M,N] f32 (+opt residual) = A[M,K] bf16 rowmajor  x  Wt[N][K] bf16.
// 128x128 macro-tile, 8 waves (2x4), wave = 64x32 (8 WMMA accs), K step 32.
// Double-buffered LDS; A tile via per-lane GLOBAL_LOAD_ASYNC_TO_LDS_B128
// (ASYNCcnt); B tile via one TDM descriptor (TENSORcnt, wave 0 issues,
// pad recreates the LDT=40 stride) when available.
// ---------------------------------------------------------------------------
#define LDT 40            // padded LDS row stride in bf16 (64B data + 16B pad)
#define TILE_E (128*LDT)  // elements per LDS tile buffer

template<int ND, int KD>
__global__ __launch_bounds__(256) void gemm_t(
    const __bf16* __restrict__ A, const __bf16* __restrict__ Wt,
    const float* __restrict__ resid, float* __restrict__ C)
{
#if USE_ASYNC_LDS
    __shared__ __bf16 As[2 * TILE_E];
    __shared__ __bf16 Bs[2 * TILE_E];
#else
    __shared__ __bf16 As[TILE_E];
    __shared__ __bf16 Bs[TILE_E];
#endif

    const int tid  = threadIdx.x;
    const int lane = tid & 31;
    const int wave = tid >> 5;
    const int wm   = wave >> 2;   // 0..1
    const int wn   = wave & 3;    // 0..3
    const int sub  = lane & 15;
    const int hi   = lane >> 4;

    const int m0 = blockIdx.y * 128;
    const int n0 = blockIdx.x * 128;

    // staging: each thread owns one 32B (16 elem) row-half of A (and B)
    const int row  = tid >> 1;    // 0..127
    const int half = tid & 1;

    const __bf16* gA = A  + (size_t)(m0 + row) * KD + 16 * half;
    const __bf16* gB = Wt + (size_t)(n0 + row) * KD + 16 * half;
    __bf16* lA = &As[row * LDT + 16 * half];
    __bf16* lB = &Bs[row * LDT + 16 * half];
    (void)gB; (void)lB;

    v8f acc[4][2];
#pragma unroll
    for (int i = 0; i < 4; ++i)
#pragma unroll
        for (int j = 0; j < 2; ++j) acc[i][j] = (v8f)0.0f;

    auto compute = [&](const __bf16* AsB, const __bf16* BsB) {
        v16bf af[4], bfr[2];
#pragma unroll
        for (int i = 0; i < 4; ++i)
            af[i] = load_fragA(AsB + (64 * wm + 16 * i + sub) * LDT, hi);
#pragma unroll
        for (int j = 0; j < 2; ++j)
            bfr[j] = load_fragB(BsB + (32 * wn + 16 * j + sub) * LDT, hi);
#pragma unroll
        for (int i = 0; i < 4; ++i)
#pragma unroll
            for (int j = 0; j < 2; ++j)
                acc[i][j] = WMMA_BF16(af[i], bfr[j], acc[i][j]);
    };

#if USE_ASYNC_LDS
    auto issueA = [&](int buf, int k0) {
        GLOBAL_AS vint4* ga = (GLOBAL_AS vint4*)(gA + k0);
        LDS_AS    vint4* la = (LDS_AS vint4*)(lA + buf * TILE_E);
        __builtin_amdgcn_global_load_async_to_lds_b128(ga, la, 0, 0);
        __builtin_amdgcn_global_load_async_to_lds_b128(ga, la, 16, 0);
    };

#if USE_TDM
    // One TDM descriptor per B tile: 128 rows x 32 bf16 (64B), row stride KD,
    // padded +16B per 64B stored -> LDS stride LDT=40 elements.
    const unsigned long long gB0  = (unsigned long long)(uintptr_t)(Wt + (size_t)n0 * KD);
    const unsigned int       ldsB = (unsigned int)(size_t)(LDS_AS __bf16*)&Bs[0];
    auto issueB = [&](int buf, int k0) {
        unsigned long long ga = gB0 + (unsigned long long)(2u * (unsigned)k0);
        vuint4 g0;
        g0[0] = 1u;                                           // count=1, user mode
        g0[1] = ldsB + (unsigned)(buf * TILE_E * 2);          // lds_addr (bytes)
        g0[2] = (unsigned)(ga & 0xFFFFFFFFu);                 // global_addr lo
        g0[3] = (unsigned)((ga >> 32) & 0x01FFFFFFu) | (2u << 30);  // addr hi | type=2
        vint8 g1;
        g1[0] = (1 << 16)        // data_size = 2B
              | (1 << 20)        // pad_enable
              | (3 << 22)        // pad_interval: 16 DWORDs (64B) between pads
              | (3 << 25);       // pad_amount: 4 DWORDs (16B)
        g1[1] = (KD & 0xFFFF) << 16;                          // tensor_dim0[15:0]
        g1[2] = (KD >> 16) | ((ND & 0xFFFF) << 16);           // dim0[31:16], dim1[15:0]
        g1[3] = (ND >> 16) | (32 << 16);                      // dim1[31:16], tile_dim0=32
        g1[4] = 128;                                          // tile_dim1=128, tile_dim2=0
        g1[5] = KD;                                           // tensor_dim0_stride[31:0]
        g1[6] = 0;                                            // stride hi, dim1_stride lo
        g1[7] = 0;
        vint4 gz4 = {0, 0, 0, 0};
        vint8 gz8 = {0, 0, 0, 0, 0, 0, 0, 0};
        __builtin_amdgcn_tensor_load_to_lds(g0, g1, gz4, gz4, gz8, 0);
    };
#else
    auto issueB = [&](int buf, int k0) {
        GLOBAL_AS vint4* gb = (GLOBAL_AS vint4*)(gB + k0);
        LDS_AS    vint4* lb = (LDS_AS vint4*)(lB + buf * TILE_E);
        __builtin_amdgcn_global_load_async_to_lds_b128(gb, lb, 0, 0);
        __builtin_amdgcn_global_load_async_to_lds_b128(gb, lb, 16, 0);
    };
#endif

    issueA(0, 0);
#if USE_TDM
    if (wave == 0) issueB(0, 0);
#else
    issueB(0, 0);
#endif
    for (int k0 = 0; k0 < KD; k0 += 32) {
        const int cur = (k0 >> 5) & 1;
        const bool more = (k0 + 32) < KD;
        if (more) {
            issueA(cur ^ 1, k0 + 32);
#if USE_TDM
            if (wave == 0) issueB(cur ^ 1, k0 + 32);
            __builtin_amdgcn_s_wait_asynccnt(2);        // tile `cur` A landed
            if (wave == 0) __builtin_amdgcn_s_wait_tensorcnt(1);  // tile `cur` B landed
#else
            issueB(cur ^ 1, k0 + 32);
            __builtin_amdgcn_s_wait_asynccnt(4);
#endif
        } else {
            __builtin_amdgcn_s_wait_asynccnt(0);
#if USE_TDM
            if (wave == 0) __builtin_amdgcn_s_wait_tensorcnt(0);
#endif
        }
        __syncthreads();                            // all waves' tile data visible
        compute(&As[cur * TILE_E], &Bs[cur * TILE_E]);
        __syncthreads();                            // safe to overwrite `cur` buf
    }
#else
    for (int k0 = 0; k0 < KD; k0 += 32) {
        bf16x8 a0 = *(const bf16x8*)(gA + k0);
        bf16x8 a1 = *(const bf16x8*)(gA + k0 + 8);
        bf16x8 b0 = *(const bf16x8*)(gB + k0);
        bf16x8 b1 = *(const bf16x8*)(gB + k0 + 8);
        *(bf16x8*)(lA)     = a0;
        *(bf16x8*)(lA + 8) = a1;
        *(bf16x8*)(lB)     = b0;
        *(bf16x8*)(lB + 8) = b1;
        if (k0 + 32 < KD) {
            __builtin_prefetch(gA + k0 + 32, 0, 1);
            __builtin_prefetch(gB + k0 + 32, 0, 1);
        }
        __syncthreads();
        compute(As, Bs);
        __syncthreads();
    }
#endif

    // ---- epilogue: C/D layout lane = column, VGPR r = row (r + 8*hi) ----
#pragma unroll
    for (int i = 0; i < 4; ++i) {
#pragma unroll
        for (int j = 0; j < 2; ++j) {
            const int n = n0 + 32 * wn + 16 * j + sub;
            const size_t cb = (size_t)(m0 + 64 * wm + 16 * i + 8 * hi) * ND + n;
#pragma unroll
            for (int r = 0; r < 8; ++r) {
                size_t idx = cb + (size_t)r * ND;
                float v = acc[i][j][r];
                if (resid) v += resid[idx];
                C[idx] = v;
            }
        }
    }
}

// ---------------------------------------------------------------------------
// Flash-style causal attention.  One wave per (b, h, 16-query tile).
// S^T = K x Q^T -> lane = query column; online softmax; O^T = V^T x P^T.
// Main loop is mask-free; only the diagonal 32-wide k-block applies the mask.
// ---------------------------------------------------------------------------
__global__ __launch_bounds__(32) void attn_kernel(
    const __bf16* __restrict__ Q, const __bf16* __restrict__ K,
    const __bf16* __restrict__ V, __bf16* __restrict__ O)
{
    __shared__ __bf16 Ps[16 * 32];   // P^T tile: [q][k_local]
    __shared__ __bf16 Vs[64 * 32];   // V^T tile: [dh][k_local]

    const int lane = threadIdx.x;
    const int sub  = lane & 15;
    const int hi   = lane >> 4;

    const int blk = blockIdx.x;          // b*H*64 + h*64 + qt
    const int qt  = blk & 63;
    const int h   = (blk >> 6) & 15;
    const int b   = blk >> 10;

    const size_t base = (size_t)b * LL * DD + (size_t)h * DH;
    const int qrow = qt * 16;

    const __bf16* qp = Q + base + (size_t)(qrow + sub) * DD;
    const v16bf bq0 = load_fragB(qp, hi);        // dh 0..31
    const v16bf bq1 = load_fragB(qp + 32, hi);   // dh 32..63

    float mrow = -3.0e38f, lrow = 0.0f;
    v8f o[4];
#pragma unroll
    for (int t = 0; t < 4; ++t) o[t] = (v8f)0.0f;

    auto process = [&](int k0, bool masked) {
        // stage V^T: lane owns global k-row k0+lane
        {
            const __bf16* vp = V + base + (size_t)(k0 + lane) * DD;
#pragma unroll
            for (int dh = 0; dh < 64; dh += 8) {
                bf16x8 v = *(const bf16x8*)(vp + dh);
#pragma unroll
                for (int e = 0; e < 8; ++e) Vs[(dh + e) * 32 + lane] = v[e];
            }
        }

        float pt[2][8];
        float tmax = -3.0e38f;
        const int qg = qrow + sub;
#pragma unroll
        for (int kt = 0; kt < 2; ++kt) {
            const int kt0 = k0 + kt * 16;
            const __bf16* kp = K + base + (size_t)(kt0 + sub) * DD;
            v16bf a0 = load_fragA(kp, hi);
            v16bf a1 = load_fragA(kp + 32, hi);
            v8f s = (v8f)0.0f;
            s = WMMA_BF16(a0, bq0, s);
            s = WMMA_BF16(a1, bq1, s);
#pragma unroll
            for (int r = 0; r < 8; ++r) {
                float sv = s[r];
                if (masked) {
                    int kg = kt0 + 8 * hi + r;
                    sv = (kg <= qg) ? sv : -3.0e38f;
                }
                pt[kt][r] = sv;
                tmax = fmaxf(tmax, sv);
            }
        }
        tmax = fmaxf(tmax, __shfl_xor(tmax, 16));

        const float mnew  = fmaxf(mrow, tmax);
        const float alpha = __expf(mrow - mnew);
        float psum = 0.0f;
#pragma unroll
        for (int kt = 0; kt < 2; ++kt) {
            bf16x8 pk;
#pragma unroll
            for (int r = 0; r < 8; ++r) {
                float p = __expf(pt[kt][r] - mnew);
                psum += p;
                pk[r] = (__bf16)p;
            }
            *(bf16x8*)&Ps[sub * 32 + kt * 16 + 8 * hi] = pk;
        }
        psum += __shfl_xor(psum, 16);
        lrow = lrow * alpha + psum;
        mrow = mnew;

#pragma unroll
        for (int t = 0; t < 4; ++t)
#pragma unroll
            for (int r = 0; r < 8; ++r) o[t][r] *= alpha;

        v16bf bp = load_fragB(&Ps[sub * 32], hi);
#pragma unroll
        for (int t = 0; t < 4; ++t) {
            v16bf av = load_fragA(&Vs[(16 * t + sub) * 32], hi);
            o[t] = WMMA_BF16(av, bp, o[t]);
        }
    };

    const int nkb = qt / 2 + 1;            // 32-wide k-blocks under causal mask
    for (int kb = 0; kb < nkb - 1; ++kb)
        process(kb * 32, false);           // fully-valid blocks: no masking
    process((nkb - 1) * 32, true);         // diagonal block: apply causal mask

    const float inv = 1.0f / lrow;
#pragma unroll
    for (int t = 0; t < 4; ++t) {
        bf16x8 pk;
#pragma unroll
        for (int r = 0; r < 8; ++r) pk[r] = (__bf16)(o[t][r] * inv);
        __bf16* op = O + base + (size_t)(qrow + sub) * DD + 16 * t + 8 * hi;
        *(bf16x8*)op = pk;
    }
}

// ---------------------------------------------------------------------------
// Elementwise kernels
// ---------------------------------------------------------------------------
__global__ __launch_bounds__(256) void embed_kernel(
    const int* __restrict__ tok, const float* __restrict__ E, float* __restrict__ Y)
{
    const int m = blockIdx.x;
    const int t = tok[m];
    const float* e = E + (size_t)t * DD;
    float* y = Y + (size_t)m * DD;
#pragma unroll
    for (int j = 0; j < 4; ++j) { int d = threadIdx.x + 256 * j; y[d] = e[d]; }
}

__global__ __launch_bounds__(256) void rmsnorm_bf16_kernel(
    const float* __restrict__ X, const float* __restrict__ g, __bf16* __restrict__ Out)
{
    const int row = blockIdx.x;
    const float* x = X + (size_t)row * DD;
    float vals[4];
    float ss = 0.0f;
#pragma unroll
    for (int j = 0; j < 4; ++j) {
        vals[j] = x[threadIdx.x + 256 * j];
        ss += vals[j] * vals[j];
    }
#pragma unroll
    for (int off = 16; off > 0; off >>= 1) ss += __shfl_xor(ss, off);
    __shared__ float red[8];
    if ((threadIdx.x & 31) == 0) red[threadIdx.x >> 5] = ss;
    __syncthreads();
    float tot = 0.0f;
#pragma unroll
    for (int w = 0; w < 8; ++w) tot += red[w];
    const float inv = rsqrtf(tot * (1.0f / (float)DD) + 1e-6f);
    __bf16* o = Out + (size_t)row * DD;
#pragma unroll
    for (int j = 0; j < 4; ++j) {
        int d = threadIdx.x + 256 * j;
        o[d] = (__bf16)(vals[j] * inv * g[d]);
    }
}

__global__ __launch_bounds__(256) void rope_cvt_kernel(
    const float* __restrict__ Qf, const float* __restrict__ Kf, const float* __restrict__ Vf,
    __bf16* __restrict__ Qb, __bf16* __restrict__ Kb, __bf16* __restrict__ Vb)
{
    const int idx = blockIdx.x * 256 + threadIdx.x;   // B*L*H*32
    const int i = idx & 31;
    const int h = (idx >> 5) & 15;
    const int l = (idx >> 9) & 1023;
    const int b = idx >> 19;
    const size_t rb = ((size_t)(b * LL + l)) * DD + (size_t)h * DH;

    const float inv_ts = __expf(-(float)i * (9.210340371976184f / 32.0f));
    const float ang = (float)l * inv_ts;
    float s, c;
    __sincosf(ang, &s, &c);

    const float q1 = Qf[rb + i], q2 = Qf[rb + i + 32];
    Qb[rb + i]      = (__bf16)((q1 * c - q2 * s) * 0.125f);
    Qb[rb + i + 32] = (__bf16)((q2 * c + q1 * s) * 0.125f);
    const float k1 = Kf[rb + i], k2 = Kf[rb + i + 32];
    Kb[rb + i]      = (__bf16)(k1 * c - k2 * s);
    Kb[rb + i + 32] = (__bf16)(k2 * c + k1 * s);
    Vb[rb + i]      = (__bf16)Vf[rb + i];
    Vb[rb + i + 32] = (__bf16)Vf[rb + i + 32];
}

__global__ __launch_bounds__(256) void swishmul_kernel(
    const float* __restrict__ G, const float* __restrict__ P, __bf16* __restrict__ Out)
{
    const size_t i = (size_t)blockIdx.x * 256 + threadIdx.x;
    const float g = G[i];
    const float p = P[i];
    Out[i] = (__bf16)((g / (1.0f + __expf(-g))) * p);
}

// ---------------------------------------------------------------------------
// Host-side launcher
// ---------------------------------------------------------------------------
template<int ND, int KD>
static inline void launch_gemm(const __bf16* A, const __bf16* Wt, const float* resid,
                               float* C, hipStream_t s)
{
    dim3 grid(ND / 128, MM / 128);
    gemm_t<ND, KD><<<grid, 256, 0, s>>>(A, Wt, resid, C);
}

static inline void transp(const float* W, __bf16* Wt, int Kd, int Nd, hipStream_t s)
{
    dim3 grid(Nd / 64, Kd / 64);
    transp_cvt_kernel<<<grid, 256, 0, s>>>(W, Wt, Kd, Nd);
}

extern "C" void kernel_launch(void* const* d_in, const int* in_sizes, int n_in,
                              void* d_out, int out_size, void* d_ws, size_t ws_size,
                              hipStream_t stream)
{
    const int*   tokens = (const int*)d_in[0];
    const float* embedW = (const float*)d_in[1];
    const float* ln1    = (const float*)d_in[2];
    const float* Wq     = (const float*)d_in[3];
    const float* Wk     = (const float*)d_in[4];
    const float* Wv     = (const float*)d_in[5];
    const float* Wo     = (const float*)d_in[6];
    const float* ln2    = (const float*)d_in[7];
    const float* Wg     = (const float*)d_in[8];
    const float* Wp     = (const float*)d_in[9];
    const float* Wd     = (const float*)d_in[10];
    const float* outln  = (const float*)d_in[11];
    const float* headW  = (const float*)d_in[12];
    float* out = (float*)d_out;

    const size_t MB = (size_t)1 << 20;
    char* ws = (char*)d_ws;
    float*  y   = (float*)(ws + 0 * MB);     //  8 MB  residual stream f32
    __bf16* xb  = (__bf16*)(ws + 8 * MB);    //  4 MB  normalized acts bf16
    float*  qf  = (float*)(ws + 12 * MB);    //  8 MB
    float*  kf  = (float*)(ws + 20 * MB);    //  8 MB
    float*  vf  = (float*)(ws + 28 * MB);    //  8 MB
    __bf16* qb  = (__bf16*)(ws + 36 * MB);   //  4 MB
    __bf16* kb  = (__bf16*)(ws + 40 * MB);   //  4 MB
    __bf16* vb  = (__bf16*)(ws + 44 * MB);   //  4 MB
    __bf16* ob  = (__bf16*)(ws + 48 * MB);   //  4 MB  attention out bf16
    float*  gf  = (float*)(ws + 52 * MB);    // 32 MB
    float*  pf  = (float*)(ws + 84 * MB);    // 32 MB
    __bf16* hb  = (__bf16*)(ws + 116 * MB);  // 16 MB
    __bf16* wqt = (__bf16*)(ws + 132 * MB);  //  2 MB  transposed bf16 weights
    __bf16* wkt = (__bf16*)(ws + 134 * MB);  //  2 MB
    __bf16* wvt = (__bf16*)(ws + 136 * MB);  //  2 MB
    __bf16* wot = (__bf16*)(ws + 138 * MB);  //  2 MB
    __bf16* wgt = (__bf16*)(ws + 140 * MB);  //  8 MB
    __bf16* wpt = (__bf16*)(ws + 148 * MB);  //  8 MB
    __bf16* wdt = (__bf16*)(ws + 156 * MB);  //  8 MB
    __bf16* hdt = (__bf16*)(ws + 164 * MB);  // 62.5 MB  transposed head

    embed_kernel<<<MM, 256, 0, stream>>>(tokens, embedW, y);

    for (int l = 0; l < NL; ++l) {
        const size_t wdd = (size_t)l * DD * DD;
        const size_t wdf = (size_t)l * DD * FF;

        // weight pre-pass: f32 [K][N] -> bf16 [N][K]
        transp(Wq + wdd, wqt, DD, DD, stream);
        transp(Wk + wdd, wkt, DD, DD, stream);
        transp(Wv + wdd, wvt, DD, DD, stream);
        transp(Wo + wdd, wot, DD, DD, stream);
        transp(Wg + wdf, wgt, DD, FF, stream);
        transp(Wp + wdf, wpt, DD, FF, stream);
        transp(Wd + (size_t)l * FF * DD, wdt, FF, DD, stream);

        rmsnorm_bf16_kernel<<<MM, 256, 0, stream>>>(y, ln1 + (size_t)l * DD, xb);

        launch_gemm<DD, DD>(xb, wqt, nullptr, qf, stream);
        launch_gemm<DD, DD>(xb, wkt, nullptr, kf, stream);
        launch_gemm<DD, DD>(xb, wvt, nullptr, vf, stream);

        rope_cvt_kernel<<<(BB * LL * HH * 32) / 256, 256, 0, stream>>>(
            qf, kf, vf, qb, kb, vb);

        attn_kernel<<<BB * HH * (LL / 16), 32, 0, stream>>>(qb, kb, vb, ob);

        // x_res = y + attn_out @ Wo   (in-place residual update of y)
        launch_gemm<DD, DD>(ob, wot, y, y, stream);

        rmsnorm_bf16_kernel<<<MM, 256, 0, stream>>>(y, ln2 + (size_t)l * DD, xb);

        launch_gemm<FF, DD>(xb, wgt, nullptr, gf, stream);
        launch_gemm<FF, DD>(xb, wpt, nullptr, pf, stream);

        swishmul_kernel<<<(MM * FF) / 256, 256, 0, stream>>>(gf, pf, hb);

        // y = y + h @ Wd
        launch_gemm<DD, FF>(hb, wdt, y, y, stream);
    }

    rmsnorm_bf16_kernel<<<MM, 256, 0, stream>>>(y, outln, xb);
    transp(headW, hdt, DD, VV, stream);
    launch_gemm<VV, DD>(xb, hdt, nullptr, out, stream);
}
